// HeteroMolAttention_48902497632378
// MI455X (gfx1250) — matrix operand order repaired and checked by reading, not compiled, not run on hardware
//
#include <hip/hip_runtime.h>
#include <hip/hip_bf16.h>

// ---------------------------------------------------------------------------
// HeteroMolAttention for MI455X (gfx1250, wave32, WMMA).
//
// GEMM: block = 128 thr (4 waves) computes a 64x64 C tile.
//   - A panel 64x32 staged per k-step into LDS as f16 in WMMA-fragment order
//     (coop float4 loads -> cvt -> ds_store_b128); waves read aligned v16h.
//   - each wave owns a 16-col strip, reuses its B fragment over 4 M-tiles
//     -> 4 v_wmma_f32_16x16x32_f16 per k-step per wave.
//   - all shapes (N,K,H) are template params: no runtime int division.
// GAT softmax/scatter: float atomics (+ monotone-uint atomicMax for seg-max).
// Workspace requirement: ~306 MB.
// ---------------------------------------------------------------------------

#define CN_MOL    40000
#define CN_FRAG   16000
#define CN_GRAPH  1024
#define CE_BOND   256000
#define CE_CROSS  64000
#define CHID      128
#define CHEADS    4
#define CHC       512

typedef __attribute__((ext_vector_type(16))) _Float16 v16h;
typedef __attribute__((ext_vector_type(8)))  _Float16 v8h;
typedef __attribute__((ext_vector_type(8)))  float    v8f;

// monotone float<->uint encoding for atomicMax on signed floats
__device__ __forceinline__ unsigned fenc(float f) {
    unsigned u = __float_as_uint(f);
    return (u & 0x80000000u) ? ~u : (u | 0x80000000u);
}
__device__ __forceinline__ float fdec(unsigned u) {
    u = (u & 0x80000000u) ? (u & 0x7FFFFFFFu) : ~u;
    return __uint_as_float(u);
}
#define ENC_NEG_INF 0x007FFFFFu   // fenc(-inf)

// ---------------------------------------------------------------- fill utils
__global__ void __launch_bounds__(256) k_fill_f32(float* p, float v, int n) {
    int t = blockIdx.x * blockDim.x + threadIdx.x;
    if (t < n) p[t] = v;
}
__global__ void __launch_bounds__(256) k_fill_u32(unsigned* p, unsigned v, int n) {
    int t = blockIdx.x * blockDim.x + threadIdx.x;
    if (t < n) p[t] = v;
}
// F is a power of two -> mask instead of %
__global__ void __launch_bounds__(256) k_bias_add(float* acc, const float* __restrict__ b,
                                                  int total, int Fmask) {
    int t = blockIdx.x * blockDim.x + threadIdx.x;
    if (t < total) acc[t] += b[t & Fmask];
}
__global__ void __launch_bounds__(256) k_elu(const float* __restrict__ in, float* out, int n) {
    int t = blockIdx.x * blockDim.x + threadIdx.x;
    if (t < n) { float x = in[t]; out[t] = x > 0.0f ? x : expm1f(x); }
}
__global__ void __launch_bounds__(256) k_relu(float* p, int n) {
    int t = blockIdx.x * blockDim.x + threadIdx.x;
    if (t < n) { float x = p[t]; p[t] = x > 0.0f ? x : 0.0f; }
}

// ------------------------------------------------------------- WMMA GEMM
// C[M,N] = act(A[M,K] @ B[K,N] + bias), optional attention epilogue:
//   aout[row*H + col/(N/H)] += raw(row,col) * avec[col]
// Block: 128 threads / 4 waves -> 64x64 tile; wave w -> cols [w*16, w*16+16).
template<int N, int K, int H, bool STORE, bool BIAS, int ACT, bool AVEC>
__global__ void __launch_bounds__(128)
k_wmma_gemm(const float* __restrict__ A, const float* __restrict__ B,
            const float* __restrict__ bias, float* __restrict__ Cst,
            const float* __restrict__ avec, float* __restrict__ aout, int M)
{
    // A panel 64 rows x 32 K, f16, stored per row in fragment-permuted order:
    //   halves [0..15]  = K {0..7, 16..23}   (lane half 0)
    //   halves [16..31] = K {8..15, 24..31}  (lane half 1)
    __shared__ __align__(64) _Float16 Alds[64 * 32];

    const int tid  = threadIdx.x;
    const int lane = tid & 31;
    const int wave = tid >> 5;

    constexpr int tilesN = N / 64;
    const int bM   = blockIdx.x / tilesN;   // constexpr divisor -> shift
    const int bN   = blockIdx.x - bM * tilesN;
    const int row0 = bM * 64;

    const int lm = lane & 15;
    const int hi = lane >> 4;
    const int col = bN * 64 + wave * 16 + lm;

    // cooperative A loader: thread -> (row in [0,64), K-half)
    const int arow  = tid >> 1;
    const int ahalf = tid & 1;
    const float* Ag = A + (size_t)(row0 + arow) * K + (ahalf ? 8 : 0);
    _Float16* Aw = &Alds[arow * 32 + ahalf * 16];

    v8f acc[4] = {{}, {}, {}, {}};

    for (int k0 = 0; k0 < K; k0 += 32) {
        // ---- stage A panel (f32 -> f16, permuted) ----
        float4 a0 = *(const float4*)(Ag + k0);
        float4 a1 = *(const float4*)(Ag + k0 + 4);
        float4 a2 = *(const float4*)(Ag + k0 + 16);
        float4 a3 = *(const float4*)(Ag + k0 + 20);
        v8h h0, h1;
        h0[0] = (_Float16)a0.x; h0[1] = (_Float16)a0.y;
        h0[2] = (_Float16)a0.z; h0[3] = (_Float16)a0.w;
        h0[4] = (_Float16)a1.x; h0[5] = (_Float16)a1.y;
        h0[6] = (_Float16)a1.z; h0[7] = (_Float16)a1.w;
        h1[0] = (_Float16)a2.x; h1[1] = (_Float16)a2.y;
        h1[2] = (_Float16)a2.z; h1[3] = (_Float16)a2.w;
        h1[4] = (_Float16)a3.x; h1[5] = (_Float16)a3.y;
        h1[6] = (_Float16)a3.z; h1[7] = (_Float16)a3.w;
        ((v8h*)Aw)[0] = h0;
        ((v8h*)Aw)[1] = h1;
        __syncthreads();

        // ---- B fragment: lane = column, half 0 -> K {0..7,16..23}, half 1 -> {8..15,24..31}
        v16h bf;
        const int kbb = k0 + (hi ? 8 : 0);
#pragma unroll
        for (int j = 0; j < 16; ++j)
            bf[j] = (_Float16)B[(size_t)(kbb + j + (j >= 8 ? 8 : 0)) * N + col];

        // ---- 4 M-tiles, reuse bf ----
#pragma unroll
        for (int t = 0; t < 4; ++t) {
            v16h af = *(const v16h*)&Alds[(t * 16 + lm) * 32 + hi * 16];
            acc[t] = __builtin_amdgcn_wmma_f32_16x16x32_f16(
                false, af, false, bf, (short)0, acc[t], false, false);
        }
        __syncthreads();
    }

    // ---- epilogue: C/D layout: vgpr i, lane half hi -> row 8*hi+i, col lm
    float bv = 0.0f, av = 0.0f;
    if constexpr (BIAS) bv = bias[col];
    if constexpr (AVEC) av = avec[col];
    constexpr int CdimShift = (N / H) == 128 ? 7 : ((N / H) == 512 ? 9 : 0);
    const int h = (H == 1) ? 0 : (col >> CdimShift);
#pragma unroll
    for (int t = 0; t < 4; ++t) {
#pragma unroll
        for (int i = 0; i < 8; ++i) {
            int row = row0 + t * 16 + hi * 8 + i;
            float raw = acc[t][i];
            if constexpr (AVEC) atomicAdd(&aout[row * H + h], raw * av);
            if constexpr (STORE) {
                float val = raw + bv;
                if constexpr (ACT == 1) val = val > 0.0f ? val : 0.01f * val;
                Cst[(size_t)row * N + col] = val;
            }
        }
    }
}

// ------------------------------------------------------------- edge kernels
template<int H>
__global__ void __launch_bounds__(256)
k_edge_attn_max(const float* __restrict__ asrc, const float* __restrict__ adst,
                const int* __restrict__ src, const int* __restrict__ dst,
                float* __restrict__ aE, unsigned* __restrict__ mU, int E)
{
    int t = blockIdx.x * blockDim.x + threadIdx.x;
    if (t >= E * H) return;
    int e = t / H, h = t - e * H;     // H constexpr -> shift/mask
    int s = src ? src[e] : e;
    int d = dst[e];
    float a = asrc[s * H + h] + adst[d * H + h];
    a = a > 0.0f ? a : 0.01f * a;     // leaky_relu(0.01)
    aE[t] = a;
    atomicMax(&mU[d * H + h], fenc(a));
}

template<int H>
__global__ void __launch_bounds__(256)
k_edge_exp_sum(const int* __restrict__ dst, const unsigned* __restrict__ mU,
               float* __restrict__ aE, float* __restrict__ sB, int E)
{
    int t = blockIdx.x * blockDim.x + threadIdx.x;
    if (t >= E * H) return;
    int e = t / H, h = t - e * H;
    int d = dst[e];
    float ev = expf(aE[t] - fdec(mU[d * H + h]));
    aE[t] = ev;
    atomicAdd(&sB[d * H + h], ev);
}

// one wave32 per edge; H*128 features per edge
template<int H>
__global__ void __launch_bounds__(256)
k_edge_scatter(const float* __restrict__ hs, const int* __restrict__ src,
               const int* __restrict__ dst, const float* __restrict__ aE,
               const float* __restrict__ sB, float* __restrict__ acc, int E)
{
    constexpr int HCl = H * 128;
    int e    = (blockIdx.x * blockDim.x + threadIdx.x) >> 5;
    int lane = threadIdx.x & 31;
    if (e >= E) return;
    int s = src ? src[e] : e;
    int d = dst[e];
#pragma unroll
    for (int f = lane; f < HCl; f += 32) {
        int h = f >> 7;
        float alpha = aE[e * H + h] / (sB[d * H + h] + 1e-16f);
        atomicAdd(&acc[(size_t)d * HCl + f], hs[(size_t)s * HCl + f] * alpha);
    }
}

// ------------------------------------------------------------- pooling / head
__global__ void __launch_bounds__(256)
k_pool_scatter(const float* __restrict__ xm, const int* __restrict__ batch,
               float* __restrict__ pool, int total)   // F = 512 hardcoded
{
    int t = blockIdx.x * blockDim.x + threadIdx.x;
    if (t >= total) return;
    int n = t >> 9, f = t & 511;
    atomicAdd(&pool[(size_t)batch[n] * 512 + f], xm[t]);
}

__global__ void __launch_bounds__(256)
k_lin2(const float* __restrict__ h, const float* __restrict__ W,
       const float* __restrict__ b, float* __restrict__ out, int G)
{
    int g    = (blockIdx.x * blockDim.x + threadIdx.x) >> 5;
    int lane = threadIdx.x & 31;
    if (g >= G) return;
    float s = 0.0f;
#pragma unroll
    for (int f = lane; f < 128; f += 32) s += h[g * 128 + f] * W[f];
#pragma unroll
    for (int o = 16; o > 0; o >>= 1) s += __shfl_xor(s, o, 32);
    if (lane == 0) out[g] = s + b[0];
}

// ---------------------------------------------------------------------------
struct GatP { const float *Wd, *Ws, *ad, *as, *b; };
struct Scratch {
    float *hs, *asr, *ads, *aE, *sB;
    unsigned* mU;
};

template<int N, int K, int H, bool STORE, bool BIAS, int ACT, bool AVEC>
static inline void gemm_launch(const float* A, const float* B, const float* bias,
                               float* Cst, const float* avec, float* aout,
                               int M, hipStream_t stream)
{
    int blocks = (M / 64) * (N / 64);
    k_wmma_gemm<N, K, H, STORE, BIAS, ACT, AVEC>
        <<<blocks, 128, 0, stream>>>(A, B, bias, Cst, avec, aout, M);
}

template<int K, int H>
static void run_gat(const float* xs, int Ns, const float* xd, int Nd,
                    const GatP& P, const int* src, const int* dst, int E,
                    float* acc, const Scratch& S, hipStream_t stream)
{
    constexpr int N = H * 128;
    k_fill_f32<<<(Ns * H + 255) / 256, 256, 0, stream>>>(S.asr, 0.0f, Ns * H);
    k_fill_f32<<<(Nd * H + 255) / 256, 256, 0, stream>>>(S.ads, 0.0f, Nd * H);
    gemm_launch<N, K, H, true,  false, 0, true>(xs, P.Ws, nullptr, S.hs, P.as, S.asr, Ns, stream);
    gemm_launch<N, K, H, false, false, 0, true>(xd, P.Wd, nullptr, nullptr, P.ad, S.ads, Nd, stream);
    k_fill_u32<<<(Nd * H + 255) / 256, 256, 0, stream>>>(S.mU, ENC_NEG_INF, Nd * H);
    k_fill_f32<<<(Nd * H + 255) / 256, 256, 0, stream>>>(S.sB, 0.0f, Nd * H);
    int eh = E * H;
    k_edge_attn_max<H><<<(eh + 255) / 256, 256, 0, stream>>>(S.asr, S.ads, src, dst, S.aE, S.mU, E);
    k_edge_exp_sum<H><<<(eh + 255) / 256, 256, 0, stream>>>(dst, S.mU, S.aE, S.sB, E);
    k_edge_scatter<H><<<(E * 32 + 255) / 256, 256, 0, stream>>>(S.hs, src, dst, S.aE, S.sB, acc, E);
    int tot = Nd * N;
    k_bias_add<<<(tot + 255) / 256, 256, 0, stream>>>(acc, P.b, tot, N - 1);
}

extern "C" void kernel_launch(void* const* d_in, const int* in_sizes, int n_in,
                              void* d_out, int out_size, void* d_ws, size_t ws_size,
                              hipStream_t stream)
{
    (void)in_sizes; (void)n_in; (void)out_size; (void)ws_size;

    // -------- inputs (insertion-ordered dict; params flattened pytree-sorted)
    const float* x_mol  = (const float*)d_in[0];
    const float* x_frag = (const float*)d_in[1];
    int p = 2;
    GatP g[2][3];   // [layer]{bond,fm,mf}{Wd,Ws,ad,as,b} (sorted keys)
    for (int l = 0; l < 2; ++l)
        for (int r = 0; r < 3; ++r) {
            g[l][r].Wd = (const float*)d_in[p++];
            g[l][r].Ws = (const float*)d_in[p++];
            g[l][r].ad = (const float*)d_in[p++];
            g[l][r].as = (const float*)d_in[p++];
            g[l][r].b  = (const float*)d_in[p++];
        }
    const float* l1f_W = (const float*)d_in[p++];
    const float* l1f_b = (const float*)d_in[p++];
    const float* l1m_W = (const float*)d_in[p++];
    const float* l1m_b = (const float*)d_in[p++];
    const float* l2_W  = (const float*)d_in[p++];
    const float* l2_b  = (const float*)d_in[p++];
    GatP mp;
    mp.Wd = (const float*)d_in[p++]; mp.Ws = (const float*)d_in[p++];
    mp.ad = (const float*)d_in[p++]; mp.as = (const float*)d_in[p++];
    mp.b  = (const float*)d_in[p++];
    const int* bond_src = (const int*)d_in[p++];
    const int* bond_dst = (const int*)d_in[p++];
    const int* mf_src   = (const int*)d_in[p++];
    const int* mf_dst   = (const int*)d_in[p++];
    const int* fm_src   = (const int*)d_in[p++];
    const int* fm_dst   = (const int*)d_in[p++];
    const int* batch    = (const int*)d_in[p++];

    // -------- workspace carve-up (floats); total ~306 MB
    float* W0 = (float*)d_ws;
    size_t off = 0;
    float* xm   = W0 + off; off += (size_t)CN_MOL  * CHC;
    float* xf   = W0 + off; off += (size_t)CN_FRAG * CHC;
    float* nm   = W0 + off; off += (size_t)CN_MOL  * CHC;
    float* nf   = W0 + off; off += (size_t)CN_FRAG * CHC;
    Scratch S;
    S.hs  = W0 + off; off += (size_t)CN_MOL * CHC;
    S.asr = W0 + off; off += (size_t)CN_MOL * CHEADS;
    S.ads = W0 + off; off += (size_t)CN_MOL * CHEADS;
    S.aE  = W0 + off; off += (size_t)CE_BOND * CHEADS;
    S.mU  = (unsigned*)(W0 + off); off += (size_t)CN_MOL * CHEADS;
    S.sB  = W0 + off; off += (size_t)CN_MOL * CHEADS;
    float* pool = W0 + off; off += (size_t)CN_GRAPH * CHC;
    float* hg   = W0 + off; off += (size_t)CN_GRAPH * CHID;

    auto fillf = [&](float* ptr, float v, int n) {
        k_fill_f32<<<(n + 255) / 256, 256, 0, stream>>>(ptr, v, n);
    };

    // -------- lin1: leaky_relu(X @ W + b)
    gemm_launch<128, 64, 1, true, true, 1, false>(x_mol,  l1m_W, l1m_b, xm, nullptr, nullptr, CN_MOL,  stream);
    gemm_launch<128, 32, 1, true, true, 1, false>(x_frag, l1f_W, l1f_b, xf, nullptr, nullptr, CN_FRAG, stream);

    // -------- hetero GAT layer 1 (K = 128)
    fillf(nm, 0.0f, CN_MOL * CHC);
    fillf(nf, 0.0f, CN_FRAG * CHC);
    run_gat<128, CHEADS>(xm, CN_MOL,  xm, CN_MOL,  g[0][0], bond_src, bond_dst, CE_BOND,  nm, S, stream);
    run_gat<128, CHEADS>(xf, CN_FRAG, xm, CN_MOL,  g[0][1], fm_src,   fm_dst,   CE_CROSS, nm, S, stream);
    run_gat<128, CHEADS>(xm, CN_MOL,  xf, CN_FRAG, g[0][2], mf_src,   mf_dst,   CE_CROSS, nf, S, stream);
    k_elu<<<(CN_MOL * CHC + 255) / 256, 256, 0, stream>>>(nm, xm, CN_MOL * CHC);
    k_elu<<<(CN_FRAG * CHC + 255) / 256, 256, 0, stream>>>(nf, xf, CN_FRAG * CHC);

    // -------- hetero GAT layer 2 (K = 512)
    fillf(nm, 0.0f, CN_MOL * CHC);
    fillf(nf, 0.0f, CN_FRAG * CHC);
    run_gat<512, CHEADS>(xm, CN_MOL,  xm, CN_MOL,  g[1][0], bond_src, bond_dst, CE_BOND,  nm, S, stream);
    run_gat<512, CHEADS>(xf, CN_FRAG, xm, CN_MOL,  g[1][1], fm_src,   fm_dst,   CE_CROSS, nm, S, stream);
    run_gat<512, CHEADS>(xm, CN_MOL,  xf, CN_FRAG, g[1][2], mf_src,   mf_dst,   CE_CROSS, nf, S, stream);
    k_elu<<<(CN_MOL * CHC + 255) / 256, 256, 0, stream>>>(nm, xm, CN_MOL * CHC);
    k_elu<<<(CN_FRAG * CHC + 255) / 256, 256, 0, stream>>>(nf, xf, CN_FRAG * CHC);

    // -------- global add pool + relu
    fillf(pool, 0.0f, CN_GRAPH * CHC);
    k_pool_scatter<<<((CN_MOL * CHC) + 255) / 256, 256, 0, stream>>>(xm, batch, pool, CN_MOL * CHC);
    k_relu<<<(CN_GRAPH * CHC + 255) / 256, 256, 0, stream>>>(pool, CN_GRAPH * CHC);

    // -------- node->graph single-head GAT (src = identity, dst = batch)
    fillf(hg, 0.0f, CN_GRAPH * CHID);
    run_gat<512, 1>(xm, CN_MOL, pool, CN_GRAPH, mp, nullptr, batch, CN_MOL, hg, S, stream);
    k_elu<<<(CN_GRAPH * CHID + 255) / 256, 256, 0, stream>>>(hg, hg, CN_GRAPH * CHID);

    // -------- lin2: [1024,128] @ [128,1] + b
    k_lin2<<<(CN_GRAPH * 32 + 255) / 256, 256, 0, stream>>>(hg, l2_W, l2_b, (float*)d_out, CN_GRAPH);
}
